// MiddleFHD_8684423872864
// MI455X (gfx1250) — matrix operand order repaired and checked by reading, not compiled, not run on hardware
//
#include <hip/hip_runtime.h>
#include <stdint.h>

// ---------------------------------------------------------------------------
// Types for CDNA5 WMMA (wave32): D(16x16,f32) = A(16x32,bf16) x B(32x16,bf16) + C
// ---------------------------------------------------------------------------
typedef __attribute__((ext_vector_type(16))) __bf16 v16bf;
typedef __attribute__((ext_vector_type(8)))  float  v8f;

union AB32 { v16bf v; uint4 q[2]; };

struct ConvParams {
  const __bf16* in;        // [D,H,W,IC]   bf16 channel-last activations
  const __bf16* wp;        // [taps][OC][IC] bf16 packed weights
  float*        out;       // [OD,OH,OW,OC] fp32 pre-BN conv output
  const uint8_t* in_mask;  // [D*H*W] 1 = empty voxel
  uint8_t*      out_mask;  // [OD*OH*OW] (written only when !subm)
  float*        stats;     // [0,OC) sum, [OC,2OC) sumsq (atomic accum)
  int D,H,W;
  int OD,OH,OW;
  int sd,sh,sw,pd,ph,pw;
  int subm;
};

__device__ __forceinline__ uint4 ldA(const __bf16* in, int D, int H, int W, int IC,
                                     int id, int ih, int iw, int c) {
  if ((unsigned)id < (unsigned)D && (unsigned)ih < (unsigned)H && (unsigned)iw < (unsigned)W)
    return *reinterpret_cast<const uint4*>(in + (((size_t)id * H + ih) * W + iw) * IC + c);
  return make_uint4(0u, 0u, 0u, 0u);
}

__device__ __forceinline__ bool occAt(const uint8_t* m, int D, int H, int W,
                                      int id, int ih, int iw) {
  if ((unsigned)id < (unsigned)D && (unsigned)ih < (unsigned)H && (unsigned)iw < (unsigned)W)
    return m[((size_t)id * H + ih) * W + iw] == 0;
  return false;
}

// ---------------------------------------------------------------------------
// Implicit-GEMM conv3d via V_WMMA_F32_16X16X32_BF16.
// One wave: 16 output voxels (M) x all OC (NT n-tiles). Fully unrolled K loop
// over taps (geometry is compile-time so all tap offsets constant-fold):
//   IC==16: two taps per K=32 block; IC==32: one tap; IC==64: two blocks/tap.
// A per-lane (16-bit A 16x32, contiguous k): lane half h -> q[0]=K[h*16..h*16+7],
//   q[1]=K[h*16+8..h*16+15]. B per-lane: lane n holds column n, K group = half.
// ---------------------------------------------------------------------------
template <int NT, int ICT, int KD, int KH, int KW>
__global__ void __launch_bounds__(256) conv_wmma(ConvParams p) {
  constexpr int OC   = NT * 16;
  constexpr int TAPS = KD * KH * KW;
  constexpr int KHW  = KH * KW;
  constexpr int NK   = (ICT == 16) ? ((TAPS + 1) >> 1)
                                   : ((ICT == 32) ? TAPS : TAPS * 2);

  __shared__ float s_stats[2 * OC];
  const int tid = threadIdx.x;
  for (int i = tid; i < 2 * OC; i += 256) s_stats[i] = 0.f;
  __syncthreads();

  const int lane = tid & 31;
  const int hseg = lane >> 4;      // which K half this lane carries
  const int mrow = lane & 15;      // A row / D column index
  const int wave = blockIdx.x * 8 + (tid >> 5);
  const int M_total = p.OD * p.OH * p.OW;
  const int tiles_m = (M_total + 15) >> 4;

  unsigned occ_bits = 0u;

  if (wave < tiles_m) {
    const int ovox = wave * 16 + mrow;
    const int ovc  = (ovox < M_total) ? ovox : (M_total - 1);
    const int ow = ovc % p.OW;
    const int t2 = ovc / p.OW;
    const int oh = t2 % p.OH;
    const int od = t2 / p.OH;
    const int ibd = od * p.sd - p.pd;
    const int ibh = oh * p.sh - p.ph;
    const int ibw = ow * p.sw - p.pw;

    // Speculative prefetch of the receptive-field center row (dropped if OOB).
    __builtin_prefetch(
        p.in + (((size_t)(ibd + KD / 2) * p.H + (ibh + KH / 2)) * p.W + (ibw + KW / 2)) * ICT,
        0, 1);

    v8f zero8 = {};
    v8f acc[NT];
#pragma unroll
    for (int n = 0; n < NT; ++n) acc[n] = zero8;

    bool occ = false;
    const uint4 z4 = make_uint4(0u, 0u, 0u, 0u);

#pragma unroll
    for (int kb = 0; kb < NK; ++kb) {
      int tapA, tapB, cA, cB;
      bool validB;
      if (ICT == 16)      { tapA = kb * 2; tapB = kb * 2 + 1; cA = 0; cB = 0; validB = (tapB < TAPS); }
      else if (ICT == 32) { tapA = kb; tapB = kb; cA = 0;  cB = 16; validB = true; }
      else                { tapA = kb >> 1; tapB = tapA; cA = (kb & 1) << 5; cB = cA + 16; validB = true; }

      const int kd = tapA / KHW, rr = tapA - kd * KHW;
      const int kh = rr / KW,    kw = rr - kh * KW;
      const int idA = ibd + kd, ihA = ibh + kh, iwA = ibw + kw;
      int idB = idA, ihB = ihA, iwB = iwA;
      if (ICT == 16 && validB) {
        const int kd2 = tapB / KHW, r2 = tapB - kd2 * KHW;
        const int kh2 = r2 / KW,    kw2 = r2 - kh2 * KW;
        idB = ibd + kd2; ihB = ibh + kh2; iwB = ibw + kw2;
      }

      AB32 a;
      a.q[0] = (hseg ? validB : true) ? ldA(p.in, p.D, p.H, p.W, ICT, hseg ? idB : idA, hseg ? ihB : ihA, hseg ? iwB : iwA, hseg ? cB : cA) : z4;
      a.q[1] = (hseg ? validB : true) ? ldA(p.in, p.D, p.H, p.W, ICT, hseg ? idB : idA, hseg ? ihB : ihA, hseg ? iwB : iwA, (hseg ? cB : cA) + 8) : z4;

      if (!p.subm) {
        const bool newTap = (ICT != 64) || ((kb & 1) == 0);
        if (newTap) {
          if (occAt(p.in_mask, p.D, p.H, p.W, idA, ihA, iwA)) occ = true;
          if (ICT == 16 && validB &&
              occAt(p.in_mask, p.D, p.H, p.W, idB, ihB, iwB)) occ = true;
        }
      }

      const int  tapH = hseg ? tapB : tapA;
      const int  cH   = hseg ? cB   : cA;
      const bool vH   = hseg ? validB : true;
      const __bf16* wb = p.wp + (size_t)tapH * OC * ICT + cH;

#pragma unroll
      for (int nt = 0; nt < NT; ++nt) {
        AB32 b;
        if (vH) {
          const __bf16* wr = wb + (size_t)(nt * 16 + mrow) * ICT;
          b.q[0] = *reinterpret_cast<const uint4*>(wr);
          b.q[1] = *reinterpret_cast<const uint4*>(wr + 8);
        } else {
          b.q[0] = z4; b.q[1] = z4;
        }
        acc[nt] = __builtin_amdgcn_wmma_f32_16x16x32_bf16(
            false, a.v, false, b.v, (short)0, acc[nt], false, false);
      }
    }

    if (!p.subm) {
      occ_bits = (unsigned)(__ballot(occ ? 1 : 0) & 0xFFFFull);
      if (hseg == 0 && ovox < M_total)
        p.out_mask[ovox] = ((occ_bits >> mrow) & 1u) ? 0 : 1;
    }

    // D layout: lanes 0-15 col=lane rows 0..7; lanes 16-31 col=lane-16 rows 8..15
#pragma unroll
    for (int nt = 0; nt < NT; ++nt) {
      const int col = nt * 16 + mrow;
#pragma unroll
      for (int r = 0; r < 8; ++r) {
        const int mo  = r + hseg * 8;
        const int vox = wave * 16 + mo;
        if (vox < M_total) {
          float val = acc[nt][r];
          const bool masked = p.subm ? (p.in_mask[vox] != 0)
                                     : (((occ_bits >> mo) & 1u) == 0u);
          if (masked) val = 0.f;
          p.out[(size_t)vox * OC + col] = val;
          atomicAdd(&s_stats[col], val);
          atomicAdd(&s_stats[OC + col], val * val);
        }
      }
    }
  }

  __syncthreads();
  for (int i = tid; i < 2 * OC; i += 256)
    atomicAdd(&p.stats[i], s_stats[i]);
}

// ---------------------------------------------------------------------------
// Helper kernels
// ---------------------------------------------------------------------------
__global__ void fill_init(uint32_t* a, size_t na, uint8_t* m, size_t nm,
                          float* s, size_t ns) {
  const size_t i      = (size_t)blockIdx.x * blockDim.x + threadIdx.x;
  const size_t stride = (size_t)gridDim.x * blockDim.x;
  for (size_t k = i; k < na; k += stride) a[k] = 0u;
  for (size_t k = i; k < nm; k += stride) m[k] = 1;
  for (size_t k = i; k < ns; k += stride) s[k] = 0.f;
}

__global__ void pack_weights(const float* __restrict__ src, __bf16* __restrict__ dst,
                             int OC, int IC, int KV) {
  const int i = blockIdx.x * blockDim.x + threadIdx.x;
  const int total = OC * IC * KV;
  if (i >= total) return;
  const int tap = i % KV;
  const int r   = i / KV;
  const int ic  = r % IC;
  const int oc  = r / IC;
  dst[((size_t)tap * OC + oc) * IC + ic] = (__bf16)src[i];
}

__global__ void scatter_voxels(const float* __restrict__ vf, const int* __restrict__ coors,
                               __bf16* __restrict__ act, uint8_t* __restrict__ mask,
                               int n, int H, int W, int C) {
  const int i = blockIdx.x * blockDim.x + threadIdx.x;
  if (i >= n) return;
  const int d = coors[i * 4 + 1];
  const int h = coors[i * 4 + 2];
  const int w = coors[i * 4 + 3];
  const size_t vox = ((size_t)d * H + h) * W + w;
  mask[vox] = 0;
  for (int c = 0; c < C; ++c) act[vox * C + c] = (__bf16)vf[i * C + c];
}

__global__ void bn_finalize(float* __restrict__ stats, const float* __restrict__ gamma,
                            const float* __restrict__ beta, int OC, float invN) {
  const int c = threadIdx.x;
  if (c >= OC) return;
  const float mean  = stats[c] * invN;
  const float var   = stats[OC + c] * invN - mean * mean;
  const float scale = gamma[c] * rsqrtf(var + 1e-5f);
  stats[2 * OC + c] = scale;
  stats[3 * OC + c] = beta[c] - mean * scale;
}

__global__ void bn_apply(const float* __restrict__ conv, const uint8_t* __restrict__ mask,
                         const float* __restrict__ stats, __bf16* __restrict__ outact,
                         float* __restrict__ finalout, int M, int OC, int OHW, int OD) {
  const int idx = blockIdx.x * blockDim.x + threadIdx.x;
  if (idx >= M * OC) return;
  const int vox = idx / OC;
  const int c   = idx - vox * OC;
  float y = 0.f;
  if (!mask[vox]) {
    const float x = conv[idx];
    y = x * stats[2 * OC + c] + stats[3 * OC + c];
    y = y > 0.f ? y : 0.f;
  }
  if (finalout) {  // last layer: write NCDHW-reshaped fp32 output [(c*OD+od), h, w]
    const int od  = vox / OHW;
    const int rem = vox - od * OHW;
    finalout[((size_t)c * OD + od) * OHW + rem] = y;
  } else {
    outact[idx] = (__bf16)y;
  }
}

// ---------------------------------------------------------------------------
// Host side
// ---------------------------------------------------------------------------
struct Cfg { int oc, ic, kd, kh, kw, sd, sh, sw, pd, ph, pw, subm; };
static const Cfg CFG[14] = {
  {16, 16, 3, 3, 3, 1, 1, 1, 1, 1, 1, 1},
  {16, 16, 3, 3, 3, 1, 1, 1, 1, 1, 1, 1},
  {32, 16, 3, 3, 3, 2, 2, 2, 1, 1, 1, 0},
  {32, 32, 3, 3, 3, 1, 1, 1, 1, 1, 1, 1},
  {32, 32, 3, 3, 3, 1, 1, 1, 1, 1, 1, 1},
  {64, 32, 3, 3, 3, 2, 2, 2, 1, 1, 1, 0},
  {64, 64, 3, 3, 3, 1, 1, 1, 1, 1, 1, 1},
  {64, 64, 3, 3, 3, 1, 1, 1, 1, 1, 1, 1},
  {64, 64, 3, 3, 3, 1, 1, 1, 1, 1, 1, 1},
  {64, 64, 3, 3, 3, 2, 2, 2, 0, 1, 1, 0},
  {64, 64, 3, 3, 3, 1, 1, 1, 1, 1, 1, 1},
  {64, 64, 3, 3, 3, 1, 1, 1, 1, 1, 1, 1},
  {64, 64, 3, 3, 3, 1, 1, 1, 1, 1, 1, 1},
  {64, 64, 3, 1, 1, 2, 1, 1, 0, 0, 0, 0},
};

extern "C" void kernel_launch(void* const* d_in, const int* in_sizes, int n_in,
                              void* d_out, int out_size, void* d_ws, size_t ws_size,
                              hipStream_t stream) {
  const float* vf    = (const float*)d_in[0];
  const int*   coors = (const int*)d_in[1];
  const int    n_vox = in_sizes[0] / 16;

  const float* wsrc[14];
  const float* gsrc[14];
  const float* bsrc[14];
  for (int i = 0; i < 14; ++i) {
    wsrc[i] = (const float*)d_in[4 + i];
    gsrc[i] = (const float*)d_in[18 + i];
    bsrc[i] = (const float*)d_in[32 + i];
  }

  // Spatial dims per layer boundary (input of layer i = index i)
  int Dd[15], Hh[15], Ww[15];
  Dd[0] = 41; Hh[0] = 200; Ww[0] = 200;
  for (int i = 0; i < 14; ++i) {
    const Cfg& c = CFG[i];
    Dd[i + 1] = (Dd[i] + 2 * c.pd - c.kd) / c.sd + 1;
    Hh[i + 1] = (Hh[i] + 2 * c.ph - c.kh) / c.sh + 1;
    Ww[i + 1] = (Ww[i] + 2 * c.pw - c.kw) / c.sw + 1;
  }
  size_t vox[15];
  int chan[15];
  chan[0] = 16;
  for (int i = 0; i < 15; ++i) vox[i] = (size_t)Dd[i] * Hh[i] * Ww[i];
  for (int i = 0; i < 14; ++i) chan[i + 1] = CFG[i].oc;

  size_t maxAct = 0, maxConv = 0, maxVox = 0;
  for (int i = 0; i < 15; ++i) {
    const size_t e = vox[i] * (size_t)chan[i];
    if (e > maxAct) maxAct = e;
    if (vox[i] > maxVox) maxVox = vox[i];
  }
  for (int i = 0; i < 14; ++i) {
    const size_t e = vox[i + 1] * (size_t)CFG[i].oc;
    if (e > maxConv) maxConv = e;
  }

  // Bump-allocate workspace
  char*  base = (char*)d_ws;
  size_t off  = 0;
  auto alloc = [&](size_t bytes) -> void* {
    void* pp = base + off;
    off = (off + bytes + 255) & ~(size_t)255;
    return pp;
  };

  __bf16* wp[14];
  for (int i = 0; i < 14; ++i) {
    const Cfg& c = CFG[i];
    const size_t taps = (size_t)c.kd * c.kh * c.kw;
    wp[i] = (__bf16*)alloc(taps * c.oc * c.ic * sizeof(__bf16));
  }
  float*   stats_all = (float*)alloc(14 * 256 * sizeof(float));
  __bf16*  actA = (__bf16*)alloc(maxAct * sizeof(__bf16));
  __bf16*  actB = (__bf16*)alloc(maxAct * sizeof(__bf16));
  float*   convbuf = (float*)alloc(maxConv * sizeof(float));
  uint8_t* maskA = (uint8_t*)alloc(maxVox);
  uint8_t* maskB = (uint8_t*)alloc(maxVox);
  (void)ws_size; (void)n_in; (void)out_size;

  // Init: zero layer-0 activations, set mask=empty, zero stats
  const size_t nzero = vox[0] * 8;  // vox0*16 bf16 = vox0*8 dwords
  const size_t ns = 14 * 256;
  const int init_blocks = (int)((nzero + 255) / 256);
  fill_init<<<init_blocks, 256, 0, stream>>>((uint32_t*)actA, nzero, maskA, vox[0],
                                             stats_all, ns);

  // Pack all weights to bf16 [tap][oc][ic]
  for (int i = 0; i < 14; ++i) {
    const Cfg& c = CFG[i];
    const int taps = c.kd * c.kh * c.kw;
    const int tot  = taps * c.oc * c.ic;
    pack_weights<<<(tot + 255) / 256, 256, 0, stream>>>(wsrc[i], wp[i], c.oc, c.ic, taps);
  }

  // Scatter voxel features
  scatter_voxels<<<(n_vox + 255) / 256, 256, 0, stream>>>(vf, coors, actA, maskA,
                                                          n_vox, Hh[0], Ww[0], 16);

  __bf16*  cur = actA; __bf16* nxt = actB;
  uint8_t* curM = maskA; uint8_t* nxtM = maskB;

  for (int i = 0; i < 14; ++i) {
    const Cfg& c = CFG[i];
    ConvParams p;
    p.in = cur; p.wp = wp[i]; p.out = convbuf;
    p.in_mask = curM;
    p.out_mask = c.subm ? nullptr : nxtM;
    p.stats = stats_all + i * 256;
    p.D = Dd[i]; p.H = Hh[i]; p.W = Ww[i];
    p.OD = Dd[i + 1]; p.OH = Hh[i + 1]; p.OW = Ww[i + 1];
    p.sd = c.sd; p.sh = c.sh; p.sw = c.sw;
    p.pd = c.pd; p.ph = c.ph; p.pw = c.pw;
    p.subm = c.subm;

    const int M_total = p.OD * p.OH * p.OW;
    const int tiles_m = (M_total + 15) / 16;
    const int blocks  = (tiles_m + 7) / 8;

    if (c.oc == 16)
      conv_wmma<1, 16, 3, 3, 3><<<blocks, 256, 0, stream>>>(p);
    else if (c.oc == 32 && c.ic == 16)
      conv_wmma<2, 16, 3, 3, 3><<<blocks, 256, 0, stream>>>(p);
    else if (c.oc == 32)
      conv_wmma<2, 32, 3, 3, 3><<<blocks, 256, 0, stream>>>(p);
    else if (c.ic == 32)
      conv_wmma<4, 32, 3, 3, 3><<<blocks, 256, 0, stream>>>(p);
    else if (c.kh == 3)
      conv_wmma<4, 64, 3, 3, 3><<<blocks, 256, 0, stream>>>(p);
    else
      conv_wmma<4, 64, 3, 1, 1><<<blocks, 256, 0, stream>>>(p);

    bn_finalize<<<1, 64, 0, stream>>>(p.stats, gsrc[i], bsrc[i], c.oc,
                                      1.f / (float)M_total);

    const uint8_t* useM = c.subm ? curM : nxtM;
    const int total = M_total * c.oc;
    const bool isLast = (i == 13);
    bn_apply<<<(total + 255) / 256, 256, 0, stream>>>(
        convbuf, useM, p.stats,
        isLast ? nullptr : nxt,
        isLast ? (float*)d_out : nullptr,
        M_total, c.oc, p.OH * p.OW, p.OD);

    __bf16* ta = cur; cur = nxt; nxt = ta;
    if (!c.subm) { uint8_t* tm = curM; curM = nxtM; nxtM = tm; }
  }
}
